// Net_74801150427731
// MI455X (gfx1250) — matrix operand order repaired and checked by reading, not compiled, not run on hardware
//
#include <hip/hip_runtime.h>

// ---------------------------------------------------------------------------
// SNN forward, fused for MI455X (gfx1250, wave32, WMMA).
//   prep_w1 : pack W1 -> f16, K padded 676->704, chunk-blocked [22][256][32]
//   fused_l1: per block: 16 batch rows x 256 hidden, loop t=0..19:
//             double-buffered X staging in LDS (f16), 22x4
//             v_wmma_f32_16x16x32_f16 (B fragments register-resident across
//             the whole t loop, courtesy of >256-VGPR gfx1250 allocation),
//             LIF recurrence in registers, spk1/mem1 written directly.
//   gemm2   : H2[T*B,16] = spk1 @ W2^T + b2 (N padded 10->16, WMMA)
//   lif2    : per-neuron scan -> spk2/mem2
// Cache policy: X and mem1/mem2 are non-temporal (stream once), spk1 stays
// regular-temporal so gemm2 re-reads it from the 192 MB L2.
// ---------------------------------------------------------------------------

typedef __attribute__((ext_vector_type(16))) _Float16 v16h;
typedef __attribute__((ext_vector_type(8)))  _Float16 v8h;
typedef __attribute__((ext_vector_type(8)))  float    v8f;

constexpr int T     = 20;
constexpr int BATCH = 8192;
constexpr int NIN   = 676;
constexpr int NKP   = 704;             // 22 * 32, zero-padded K
constexpr int NHID  = 256;
constexpr int NOUT  = 10;
constexpr int M1    = T * BATCH;       // 163840
constexpr int KC1   = NKP / 32;        // 22 K-chunks

// ---------------------------------------------------------------------------
// Pack W1 (f32 [256][676]) -> W1b (f16 [22][256][32], zero-padded K).
// ---------------------------------------------------------------------------
__global__ __launch_bounds__(256)
void prep_w1(const float* __restrict__ W1, _Float16* __restrict__ W1b)
{
    const int i = blockIdx.x * 256 + threadIdx.x;
    if (i >= KC1 * NHID * 32) return;
    const int kk = i & 31;
    const int n  = (i >> 5) & 255;
    const int kc = i >> 13;            // 256*32 = 8192 = 2^13
    const int k  = kc * 32 + kk;
    W1b[i] = (k < NIN) ? (_Float16)W1[(size_t)n * NIN + k] : (_Float16)0.0f;
}

// ---------------------------------------------------------------------------
// Stage one 16x676 X row-tile into LDS as f16, zero-padded to 704 columns.
// Non-temporal loads: X is read exactly once, keep it out of L2.
// ---------------------------------------------------------------------------
__device__ __forceinline__
void stage_xtile(_Float16 (&dst)[16][NKP], const float* __restrict__ Xrow,
                 int srow, int scol0)
{
    for (int c = scol0; c < NKP; c += 32) {
        if (c < NIN) {                 // NIN % 4 == 0: whole slice valid
            #pragma unroll
            for (int u = 0; u < 4; ++u)
                dst[srow][c + u] =
                    (_Float16)__builtin_nontemporal_load(Xrow + c + u);
        } else {
            #pragma unroll
            for (int u = 0; u < 4; ++u) dst[srow][c + u] = (_Float16)0.0f;
        }
    }
}

// ---------------------------------------------------------------------------
// Fused layer-1 GEMM + LIF scan.
// Grid: BATCH/16 = 512 blocks of 128 threads (4 waves).
// Wave w owns N-tiles [w*4, w*4+4) of the 16x256 output strip.
// ---------------------------------------------------------------------------
__global__ __launch_bounds__(128)
void fused_l1(const float* __restrict__ X, const _Float16* __restrict__ W1b,
              const float* __restrict__ b1,
              float* __restrict__ spk1, float* __restrict__ mem1)
{
    __shared__ __align__(16) _Float16 As[2][16][NKP];   // 45 KB ping-pong

    const int bBase  = blockIdx.x * 16;
    const int tid    = threadIdx.x;
    const int wave   = tid >> 5;          // 0..3 -> N group
    const int lane   = tid & 31;
    const int lrow   = lane & 15;         // M row / N col inside 16x16 tile
    const int hiHalf = lane >> 4;         // lanes 0..15 vs 16..31

    float bias[4];
    #pragma unroll
    for (int tn = 0; tn < 4; ++tn)
        bias[tn] = b1[(wave * 4 + tn) * 16 + lrow];

    // LDS staging assignment: thread -> (row, 4-float column slice)
    const int srow  = tid >> 3;           // 0..15
    const int scol0 = (tid & 7) * 4;      // 0,4,...,28

    v8f ms[4] = {};                       // membrane potential (C layout)

    // prologue: stage t = 0
    stage_xtile(As[0], X + ((size_t)bBase + srow) * NIN, srow, scol0);
    __syncthreads();

    #pragma unroll 1
    for (int t = 0; t < T; ++t) {
        const int cur = t & 1;

        // ---- prefetch t+1 into the idle buffer; overlaps with compute ----
        if (t + 1 < T) {
            const float* Xn =
                X + ((size_t)(t + 1) * BATCH + bBase + srow) * NIN;
            stage_xtile(As[cur ^ 1], Xn, srow, scol0);
        }

        // ---- 22-chunk K loop, 4 WMMAs per chunk ----
        v8f acc[4] = {};
        #pragma unroll
        for (int kc = 0; kc < KC1; ++kc) {
            // A fragment: lane<16 -> K 0..7 / 16..23 ; lane>=16 -> +8
            const int ak = kc * 32 + hiHalf * 8;
            v8h alo = *(const v8h*)&As[cur][lrow][ak];
            v8h ahi = *(const v8h*)&As[cur][lrow][ak + 16];
            v16h afrag;
            #pragma unroll
            for (int u = 0; u < 8; ++u) { afrag[u] = alo[u]; afrag[u + 8] = ahi[u]; }

            // B fragments from packed W1b (register/L2 resident)
            const _Float16* wb = W1b + (size_t)kc * (NHID * 32);
            #pragma unroll
            for (int tn = 0; tn < 4; ++tn) {
                const int col = (wave * 4 + tn) * 16 + lrow;
                const _Float16* bp = wb + col * 32 + hiHalf * 8;
                v8h b0  = *(const v8h*)bp;
                v8h b1v = *(const v8h*)(bp + 16);
                v16h bfrag;
                #pragma unroll
                for (int u = 0; u < 8; ++u) { bfrag[u] = b0[u]; bfrag[u + 8] = b1v[u]; }
                acc[tn] = __builtin_amdgcn_wmma_f32_16x16x32_f16(
                    false, afrag, false, bfrag, (short)0, acc[tn], false, false);
            }
        }

        // ---- LIF update in registers; C layout: M = r + 8*hiHalf, N = lrow
        const int rowBase = bBase + hiHalf * 8;
        #pragma unroll
        for (int tn = 0; tn < 4; ++tn) {
            const int col = (wave * 4 + tn) * 16 + lrow;
            #pragma unroll
            for (int r = 0; r < 8; ++r) {
                float m = ms[tn][r];
                m = 0.9f * m + (acc[tn][r] + bias[tn]);
                const float s = (m > 1.0f) ? 1.0f : 0.0f;   // spike(m - THR)
                m -= s;                                     // soft reset
                ms[tn][r] = m;
                const size_t idx = ((size_t)t * BATCH + rowBase + r) * NHID + col;
                spk1[idx] = s;                              // re-read by gemm2: RT
                __builtin_nontemporal_store(m, &mem1[idx]); // write-only: NT
            }
        }
        __syncthreads();   // staging of t+1 done; As[cur] free for t+2
    }
}

// ---------------------------------------------------------------------------
// GEMM2: H2[M1,16] = spk1[M1,256] @ W2^T (+ b2), N padded 10->16.
// Block: 128 threads (4 waves), tile 64M x 16N, K = 8 chunks of 32.
// ---------------------------------------------------------------------------
__global__ __launch_bounds__(128)
void gemm2_wmma(const float* __restrict__ S1, const float* __restrict__ W2,
                const float* __restrict__ b2, float* __restrict__ H2)
{
    __shared__ __align__(16) _Float16 As[64][32];
    __shared__ __align__(16) _Float16 Bs[16][256];   // [n][k], n>=10 zero pad

    const int mBase  = blockIdx.x * 64;
    const int tid    = threadIdx.x;
    const int wave   = tid >> 5;
    const int lane   = tid & 31;
    const int lrow   = lane & 15;
    const int hiHalf = lane >> 4;

    for (int i = tid; i < 16 * 256; i += 128) {
        const int n = i >> 8, k = i & 255;
        Bs[n][k] = (n < NOUT) ? (_Float16)W2[n * 256 + k] : (_Float16)0.0f;
    }

    v8f acc = {};
    for (int kc = 0; kc < NHID / 32; ++kc) {
        const int row = tid >> 1;
        const int ks  = (tid & 1) << 4;
        const float* src = S1 + (size_t)(mBase + row) * NHID + kc * 32 + ks;
        #pragma unroll
        for (int u = 0; u < 16; ++u)
            As[row][ks + u] = (_Float16)__builtin_nontemporal_load(src + u);
        __syncthreads();

        const int arow = wave * 16 + lrow;
        const int ak   = hiHalf * 8;
        v8h alo = *(const v8h*)&As[arow][ak];
        v8h ahi = *(const v8h*)&As[arow][ak + 16];
        v16h afrag;
        #pragma unroll
        for (int u = 0; u < 8; ++u) { afrag[u] = alo[u]; afrag[u + 8] = ahi[u]; }

        const _Float16* bp = &Bs[lrow][kc * 32 + hiHalf * 8];
        v8h b0  = *(const v8h*)bp;
        v8h b1v = *(const v8h*)(bp + 16);
        v16h bfrag;
        #pragma unroll
        for (int u = 0; u < 8; ++u) { bfrag[u] = b0[u]; bfrag[u + 8] = b1v[u]; }

        acc = __builtin_amdgcn_wmma_f32_16x16x32_f16(
            false, afrag, false, bfrag, (short)0, acc, false, false);
        __syncthreads();
    }

    const int crow0 = mBase + wave * 16 + hiHalf * 8;
    const float bias = (lrow < NOUT) ? b2[lrow] : 0.0f;
    #pragma unroll
    for (int r = 0; r < 8; ++r)
        H2[(size_t)(crow0 + r) * 16 + lrow] = acc[r] + bias;
}

// ---------------------------------------------------------------------------
// LIF scan layer 2: one thread per (batch, out) neuron.
// ---------------------------------------------------------------------------
__global__ __launch_bounds__(256)
void lif2_scan(const float* __restrict__ H2,
               float* __restrict__ spk2, float* __restrict__ mem2)
{
    const int i = blockIdx.x * blockDim.x + threadIdx.x;
    if (i >= BATCH * NOUT) return;
    const int b = i / NOUT, o = i % NOUT;
    float m = 0.0f;
    #pragma unroll
    for (int t = 0; t < T; ++t) {
        const float h = H2[(size_t)t * (BATCH * 16) + b * 16 + o];
        m = 0.9f * m + h;
        const float s = (m > 1.0f) ? 1.0f : 0.0f;
        m -= s;
        const size_t idx = (size_t)t * (BATCH * NOUT) + i;
        __builtin_nontemporal_store(s, &spk2[idx]);
        __builtin_nontemporal_store(m, &mem2[idx]);
    }
}

// ---------------------------------------------------------------------------
// Host launcher.  d_in: x, W1, b1, W2, b2.  d_out (flat f32, in order):
// spk2[T,B,10], spk1[T,B,256], mem2[T,B,10], mem1[T,B,256].
// Workspace: W1b f16 (352 KB) at offset 0, H2 f32 (10.5 MB) at 512 KB.
// ---------------------------------------------------------------------------
extern "C" void kernel_launch(void* const* d_in, const int* in_sizes, int n_in,
                              void* d_out, int out_size, void* d_ws, size_t ws_size,
                              hipStream_t stream)
{
    const float* x  = (const float*)d_in[0];
    const float* W1 = (const float*)d_in[1];
    const float* b1 = (const float*)d_in[2];
    const float* W2 = (const float*)d_in[3];
    const float* b2 = (const float*)d_in[4];

    float* out  = (float*)d_out;
    float* spk2 = out;
    float* spk1 = spk2 + (size_t)T * BATCH * NOUT;
    float* mem2 = spk1 + (size_t)T * BATCH * NHID;
    float* mem1 = mem2 + (size_t)T * BATCH * NOUT;

    _Float16* W1b = (_Float16*)d_ws;                       // [22][256][32]
    float*    H2  = (float*)((char*)d_ws + (512 * 1024));  // [M1][16]

    prep_w1<<<(KC1 * NHID * 32 + 255) / 256, 256, 0, stream>>>(W1, W1b);

    fused_l1<<<BATCH / 16, 128, 0, stream>>>(x, W1b, b1, spk1, mem1);

    gemm2_wmma<<<M1 / 64, 128, 0, stream>>>(spk1, W2, b2, H2);

    lif2_scan<<<(BATCH * NOUT + 255) / 256, 256, 0, stream>>>(H2, spk2, mem2);
}
